// TriangleMultiplication_10806137717295
// MI455X (gfx1250) — matrix operand order repaired and checked by reading, not compiled, not run on hardware
//
#include <hip/hip_runtime.h>
#include <cstdint>
#include <cstddef>

// ---------------------------------------------------------------------------
// Triangle multiplication (outgoing) for MI455X / gfx1250.
// Bandwidth-bound problem (~77 GFLOP vs ~0.6 GB traffic @ 23.3 TB/s), so all
// GEMMs run as bf16 v_wmma_f32_16x16x32_bf16 (f32 accum), staged tensors are
// bf16, and the triangle einsum operands are pre-permuted to [d][i][k] /
// [d][k][j] so each of the 128 per-channel GEMMs is contiguous.
// B operands are kept in WMMA fragment order (weights pre-permuted in global;
// triangle B tiles transposed during the LDS stage) so every operand load is
// a b128, not a 16-element gather.
// ---------------------------------------------------------------------------

typedef __attribute__((ext_vector_type(16))) __bf16 v16bf;
typedef __attribute__((ext_vector_type(8)))  float  v8f;

#define DEVI static __device__ __forceinline__

constexpr int   kN  = 512;
constexpr int   kD  = 128;
constexpr int   kNP = kN * kN;          // 262144 pixels
constexpr float kKeep  = 0.75f;
constexpr float kLnEps = 1e-5f;

struct BF16x16 { unsigned short u[16]; };

DEVI unsigned short f2bf(float f) {
    unsigned int u = __builtin_bit_cast(unsigned int, f);
    unsigned int r = u + 0x7FFFu + ((u >> 16) & 1u);   // round-nearest-even
    return (unsigned short)(r >> 16);
}
DEVI float bf2f(unsigned short h) {
    unsigned int u = ((unsigned int)h) << 16;
    return __builtin_bit_cast(float, u);
}
DEVI float sigmoidf(float x) { return 1.0f / (1.0f + __expf(-x)); }

// Load a 16x32 bf16 WMMA operand as two 16B chunks at base+off and
// base+off+step (step=16 for A from row-major [row][k] memory where the
// lane-half's second K block sits 16 elements later; step=8 for fragments
// stored contiguously per lane).
DEVI v16bf loadPair(const unsigned short* __restrict__ base, size_t off, int step) {
    BF16x16 t;
    *reinterpret_cast<uint4*>(&t.u[0]) = *reinterpret_cast<const uint4*>(base + off);
    *reinterpret_cast<uint4*>(&t.u[8]) = *reinterpret_cast<const uint4*>(base + off + step);
    return __builtin_bit_cast(v16bf, t);
}

DEVI v8f wmma_bf16(v16bf a, v16bf b, v8f c) {
    return __builtin_amdgcn_wmma_f32_16x16x32_bf16(
        false, a, false, b, (short)0, c, false, false);
}

// ---------------------------------------------------------------------------
// K0: weights fp32 [o][d] -> bf16 in WMMA B-fragment order:
//   wF[q][kt][nt][lane][e], e=0..15,  k = kt*32 + (e<8 ? (lane>>4)*8 + e : 16 + (lane>>4)*8 + e-8),
//                                     n = nt*16 + (lane&15)
// so each lane's 32-byte fragment is contiguous (loadPair step=8).
// Mat order q: gate_a, proj_a, gate_b, proj_b, gate_out, proj_out.
// ---------------------------------------------------------------------------
__global__ void kConvertWeights(const float* w0, const float* w1, const float* w2,
                                const float* w3, const float* w4, const float* w5,
                                unsigned short* __restrict__ wF) {
    const float* src[6] = {w0, w1, w2, w3, w4, w5};
    int idx = blockIdx.x * blockDim.x + threadIdx.x;
    if (idx >= 6 * kD * kD) return;
    int e    = idx & 15;
    int lane = (idx >> 4) & 31;
    int nt   = (idx >> 9) & 7;
    int kt   = (idx >> 12) & 3;
    int q    = idx >> 14;
    int k = kt * 32 + (e < 8 ? (lane >> 4) * 8 + e : 16 + (lane >> 4) * 8 + (e - 8));
    int n = nt * 16 + (lane & 15);
    wF[idx] = f2bf(src[q][n * kD + k]);
}

// ---------------------------------------------------------------------------
// K1/K6: LayerNorm over D=128, one wave32 per pixel, bf16 output.
// ---------------------------------------------------------------------------
__global__ void kLayerNorm(const float* __restrict__ src,
                           const float* __restrict__ gamma,
                           const float* __restrict__ beta,
                           unsigned short* __restrict__ dst) {
    int wave = threadIdx.x >> 5, lane = threadIdx.x & 31;
    size_t p = (size_t)blockIdx.x * 8 + wave;
    const float4 v = *reinterpret_cast<const float4*>(src + p * kD + lane * 4);
    float s  = v.x + v.y + v.z + v.w;
    float sq = v.x * v.x + v.y * v.y + v.z * v.z + v.w * v.w;
#pragma unroll
    for (int o = 16; o >= 1; o >>= 1) {
        s  += __shfl_xor(s,  o, 32);
        sq += __shfl_xor(sq, o, 32);
    }
    float mean = s * (1.0f / kD);
    float var  = sq * (1.0f / kD) - mean * mean;
    float rstd = rsqrtf(var + kLnEps);
    int c = lane * 4;
    float y0 = (v.x - mean) * rstd * gamma[c + 0] + beta[c + 0];
    float y1 = (v.y - mean) * rstd * gamma[c + 1] + beta[c + 1];
    float y2 = (v.z - mean) * rstd * gamma[c + 2] + beta[c + 2];
    float y3 = (v.w - mean) * rstd * gamma[c + 3] + beta[c + 3];
    uint2 pk;
    pk.x = (unsigned)f2bf(y0) | ((unsigned)f2bf(y1) << 16);
    pk.y = (unsigned)f2bf(y2) | ((unsigned)f2bf(y3) << 16);
    *reinterpret_cast<uint2*>(dst + p * kD + c) = pk;
}

// ---------------------------------------------------------------------------
// K2: fused 5 linears + gating + per-pixel norm^2. One wave per 16-pixel tile;
// all 5 bf16 fragment-ordered weight mats staged in 160 KB LDS (WGP: 320 KB).
// 160 v_wmma_f32_16x16x32_bf16 per wave; every B load is 2x ds_load_b128.
// ---------------------------------------------------------------------------
__global__ void kLinears(const unsigned short* __restrict__ m,
                         const unsigned short* __restrict__ wF,
                         unsigned short* __restrict__ aB,
                         unsigned short* __restrict__ bB,
                         unsigned short* __restrict__ gateB,
                         float* __restrict__ nsqA,
                         float* __restrict__ nsqB) {
    extern __shared__ unsigned short s_w[];   // 5 * 128 * 128 bf16 = 160 KB
    {
        const uint4* src = reinterpret_cast<const uint4*>(wF);
        uint4* dst = reinterpret_cast<uint4*>(s_w);
        const int nvec = 5 * kD * kD / 8;     // 10240 x 16B
        for (int i = threadIdx.x; i < nvec; i += blockDim.x) dst[i] = src[i];
    }
    __syncthreads();

    int wave = threadIdx.x >> 5, lane = threadIdx.x & 31;
    int g = lane >> 4, col = lane & 15, row = lane & 15;
    int p0 = (blockIdx.x * 8 + wave) * 16;

    v16bf A[4];
#pragma unroll
    for (int kt = 0; kt < 4; ++kt)
        A[kt] = loadPair(m, (size_t)(p0 + row) * kD + kt * 32 + g * 8, 16);

    float na[8], nb[8];
#pragma unroll
    for (int r = 0; r < 8; ++r) { na[r] = 0.f; nb[r] = 0.f; }

    for (int nt = 0; nt < 8; ++nt) {
        v8f acc[5];
#pragma unroll
        for (int q = 0; q < 5; ++q) acc[q] = v8f{};
#pragma unroll
        for (int q = 0; q < 5; ++q)
#pragma unroll
            for (int kt = 0; kt < 4; ++kt)
                acc[q] = wmma_bf16(
                    A[kt],
                    loadPair(s_w, ((size_t)(q * 32 + kt * 8 + nt) << 9) + lane * 16, 8),
                    acc[q]);
#pragma unroll
        for (int r = 0; r < 8; ++r) {
            size_t idx = (size_t)(p0 + r + g * 8) * kD + nt * 16 + col;
            float av = sigmoidf(acc[0][r]) * acc[1][r];
            float bv = sigmoidf(acc[2][r]) * acc[3][r];
            float gv = sigmoidf(acc[4][r]);
            aB[idx] = f2bf(av);
            bB[idx] = f2bf(bv);
            gateB[idx] = f2bf(gv);
            na[r] += av * av;
            nb[r] += bv * bv;
        }
    }
    // reduce over the 16 lanes sharing each row (xor stays inside lane-half)
#pragma unroll
    for (int o = 1; o < 16; o <<= 1)
#pragma unroll
        for (int r = 0; r < 8; ++r) {
            na[r] += __shfl_xor(na[r], o, 32);
            nb[r] += __shfl_xor(nb[r], o, 32);
        }
    if (col == 0) {
        int base = p0 + g * 8;
#pragma unroll
        for (int r = 0; r < 8; ++r) {
            nsqA[base + r] = na[r];
            nsqB[base + r] = nb[r];
        }
    }
}

// ---------------------------------------------------------------------------
// K3: quantile threshold via float-bit-monotonic histogram (norms >= 0).
// ---------------------------------------------------------------------------
__global__ void kZeroHist(int* hist) {
    int i = blockIdx.x * blockDim.x + threadIdx.x;
    if (i < 2 * 65536) hist[i] = 0;
}
__global__ void kHist(const float* __restrict__ na, const float* __restrict__ nb,
                      int* __restrict__ hist) {
    int i = blockIdx.x * blockDim.x + threadIdx.x;
    if (i >= kNP) return;
    unsigned int ba = __builtin_bit_cast(unsigned int, na[i]) >> 15;
    unsigned int bb = __builtin_bit_cast(unsigned int, nb[i]) >> 15;
    if (ba > 65535u) ba = 65535u;
    if (bb > 65535u) bb = 65535u;
    atomicAdd(&hist[ba], 1);
    atomicAdd(&hist[65536 + bb], 1);
}
__global__ void kQuantile(const int* __restrict__ hist, float* __restrict__ thr) {
    __shared__ int chunk[256];
    for (int h = 0; h < 2; ++h) {
        const int* hh = hist + h * 65536;
        int t = threadIdx.x, s = 0;
        for (int i = 0; i < 256; ++i) s += hh[t * 256 + i];
        chunk[t] = s;
        __syncthreads();
        if (t == 0) {
            const int target = (int)((1.0f - kKeep) * kNP);   // 65536
            int cum = 0, bin = 65535;
            for (int c = 0; c < 256; ++c) {
                if (cum + chunk[c] >= target) {
                    int local = cum;
                    for (int i = 0; i < 256; ++i) {
                        local += hh[c * 256 + i];
                        if (local >= target) { bin = c * 256 + i; break; }
                    }
                    break;
                }
                cum += chunk[c];
            }
            unsigned int bits = ((unsigned int)(bin + 1)) << 15;
            thr[h] = __builtin_bit_cast(float, bits);   // threshold on norm^2
        }
        __syncthreads();
    }
}

// ---------------------------------------------------------------------------
// K4: masked/scaled LDS-tiled permute to GEMM layouts.
//   a[i][k][d] -> a_t[d][i*512+k]   (inOffMulZ=65536, inRowStride=128, pR=1,  pZ=512)
//   b[j][k][d] -> b_t[d][k*512+j]   (inOffMulZ=128,   inRowStride=65536, pR=512, pZ=1)
// ---------------------------------------------------------------------------
__global__ void kPermute(const unsigned short* __restrict__ in,
                         unsigned short* __restrict__ out,
                         const float* __restrict__ nsq,
                         const float* __restrict__ thr,
                         int inOffMulZ, int inRowStride, int pMulR, int pMulZ) {
    __shared__ unsigned short tile[32][33];
    int z = blockIdx.z;
    int r0 = blockIdx.x * 32, c0 = blockIdx.y * 32;
    float th = *thr;
    const float scale = 1.0f / (kKeep + 1e-8f);
#pragma unroll
    for (int s = 0; s < 4; ++s) {
        int r = r0 + threadIdx.y + s * 8;
        int c = c0 + threadIdx.x;
        float v = bf2f(in[(size_t)z * inOffMulZ + (size_t)r * inRowStride + c]);
        int p = r * pMulR + z * pMulZ;
        v = (nsq[p] >= th) ? v * scale : 0.0f;
        tile[threadIdx.y + s * 8][threadIdx.x] = f2bf(v);
    }
    __syncthreads();
#pragma unroll
    for (int s = 0; s < 4; ++s) {
        int c = c0 + threadIdx.y + s * 8;
        int r = r0 + threadIdx.x;
        out[(size_t)c * kNP + (size_t)z * kN + r] = tile[threadIdx.x][threadIdx.y + s * 8];
    }
}

// ---------------------------------------------------------------------------
// K5: triangle einsum. For each d: T_d = A_d(512x512) @ B_d(512x512), bf16
// WMMA with f32 accum. Block = 8 waves sharing one j-tile; the 32x16 B tile
// is staged TRANSPOSED in LDS (Bt[j_local][k_local], pitch 40 -> conflict-free
// and 16B aligned) so each lane's fragment is two ds_load_b128s.
// A rows prefetched (global_prefetch_b8).
// ---------------------------------------------------------------------------
constexpr int kBtPitch = 40;   // elems; 80B rows: 16B aligned, conflict-free

__global__ void kTriangle(const unsigned short* __restrict__ at,
                          const unsigned short* __restrict__ btg,
                          float* __restrict__ tout) {
    __shared__ unsigned short Bt[16 * kBtPitch];
    int wave = threadIdx.x >> 5, lane = threadIdx.x & 31;
    int g = lane >> 4, col = lane & 15, row = lane & 15;
    int j0 = blockIdx.x * 16;
    int i0 = (blockIdx.y * 8 + wave) * 16;
    int d  = blockIdx.z;
    const unsigned short* aP = at  + (size_t)d * kNP;
    const unsigned short* bP = btg + (size_t)d * kNP;

    v8f acc = v8f{};
    for (int s = 0; s < 16; ++s) {
        int k0 = s * 32;
        __syncthreads();
        {   // cooperative transposed stage of the shared 32x16 B tile
            int t = threadIdx.x, kk = t >> 3, jj = (t & 7) * 2;
            unsigned int v =
                *reinterpret_cast<const unsigned int*>(&bP[(size_t)(k0 + kk) * kN + j0 + jj]);
            Bt[jj * kBtPitch + kk]       = (unsigned short)(v & 0xFFFFu);
            Bt[(jj + 1) * kBtPitch + kk] = (unsigned short)(v >> 16);
        }
        __syncthreads();
        size_t aoff = (size_t)(i0 + row) * kN + k0 + g * 8;
        if (s + 1 < 16)
            __builtin_prefetch(aP + aoff + 32, 0, 1);
        v16bf Aa = loadPair(aP, aoff, 16);
        v16bf Bb = loadPair(Bt, (size_t)col * kBtPitch + g * 8, 16);
        acc = wmma_bf16(Aa, Bb, acc);
    }
#pragma unroll
    for (int r = 0; r < 8; ++r)
        tout[((size_t)(i0 + r + g * 8) * kN + j0 + col) * kD + d] = acc[r];
}

// ---------------------------------------------------------------------------
// K7: out = z + sigmoid_gate * (LN(t) @ Wproj_out^T). 32 WMMAs per wave;
// weight fragments (mat 5) read straight from global as 2x global_load_b128.
// ---------------------------------------------------------------------------
__global__ void kFinal(const unsigned short* __restrict__ tln,
                       const unsigned short* __restrict__ wF,
                       const unsigned short* __restrict__ gate,
                       const float* __restrict__ z,
                       float* __restrict__ out) {
    int wave = threadIdx.x >> 5, lane = threadIdx.x & 31;
    int g = lane >> 4, col = lane & 15, row = lane & 15;
    int p0 = (blockIdx.x * 8 + wave) * 16;
    v16bf A[4];
#pragma unroll
    for (int kt = 0; kt < 4; ++kt)
        A[kt] = loadPair(tln, (size_t)(p0 + row) * kD + kt * 32 + g * 8, 16);
    for (int nt = 0; nt < 8; ++nt) {
        v8f acc = v8f{};
#pragma unroll
        for (int kt = 0; kt < 4; ++kt)
            acc = wmma_bf16(
                A[kt],
                loadPair(wF, ((size_t)(5 * 32 + kt * 8 + nt) << 9) + lane * 16, 8),
                acc);
#pragma unroll
        for (int r = 0; r < 8; ++r) {
            size_t idx = (size_t)(p0 + r + g * 8) * kD + nt * 16 + col;
            out[idx] = z[idx] + bf2f(gate[idx]) * acc[r];
        }
    }
}

// ---------------------------------------------------------------------------
// Host launcher. Workspace layout with lifetime-based aliasing (~339 MB):
//   wF | m(->a_t->tln) | aB+bB(->t fp32) | gate | b_t | nsqA | nsqB | hist | thr
// ---------------------------------------------------------------------------
extern "C" void kernel_launch(void* const* d_in, const int* in_sizes, int n_in,
                              void* d_out, int out_size, void* d_ws, size_t ws_size,
                              hipStream_t stream) {
    (void)in_sizes; (void)n_in; (void)out_size; (void)ws_size;
    const float* z         = (const float*)d_in[0];
    const float* ln_in_g   = (const float*)d_in[1];
    const float* ln_in_b   = (const float*)d_in[2];
    const float* w_proj_a  = (const float*)d_in[3];
    const float* w_gate_a  = (const float*)d_in[4];
    const float* w_proj_b  = (const float*)d_in[5];
    const float* w_gate_b  = (const float*)d_in[6];
    const float* w_gate_o  = (const float*)d_in[7];
    const float* ln_out_g  = (const float*)d_in[8];
    const float* ln_out_b  = (const float*)d_in[9];
    const float* w_proj_o  = (const float*)d_in[10];
    float* out = (float*)d_out;

    char* ws = (char*)d_ws;
    const size_t szBF = (size_t)kNP * kD * sizeof(unsigned short);  // 64 MB
    size_t off = 0;
    unsigned short* wF   = (unsigned short*)(ws + off); off += 6 * kD * kD * sizeof(unsigned short);
    off = (off + 255) & ~(size_t)255;
    unsigned short* mB   = (unsigned short*)(ws + off); off += szBF;   // -> a_t -> tln
    unsigned short* aB   = (unsigned short*)(ws + off); off += szBF;   // \  -> t (fp32,
    unsigned short* bB   = (unsigned short*)(ws + off); off += szBF;   // /     128 MB)
    unsigned short* gateB= (unsigned short*)(ws + off); off += szBF;
    unsigned short* b_t  = (unsigned short*)(ws + off); off += szBF;
    float* nsqA = (float*)(ws + off); off += (size_t)kNP * sizeof(float);
    float* nsqB = (float*)(ws + off); off += (size_t)kNP * sizeof(float);
    int*   hist = (int*)(ws + off);   off += 2 * 65536 * sizeof(int);
    float* thr  = (float*)(ws + off); off += 256;
    unsigned short* a_t = mB;          // m dead after kLinears
    unsigned short* tln = mB;          // a_t dead after kTriangle
    float* tF = (float*)aB;            // aB+bB dead after permutes (128 MB)

    // K0: bf16 fragment-ordered weights
    kConvertWeights<<<(6 * kD * kD + 255) / 256, 256, 0, stream>>>(
        w_gate_a, w_proj_a, w_gate_b, w_proj_b, w_gate_o, w_proj_o, wF);
    // K1: LN(z) -> m bf16
    kLayerNorm<<<kNP / 8, 256, 0, stream>>>(z, ln_in_g, ln_in_b, mB);
    // K2: 5 fused WMMA linears + gating + norms (160 KB dynamic LDS)
    kLinears<<<kNP / 128, 256, 5 * kD * kD * sizeof(unsigned short), stream>>>(
        mB, wF, aB, bB, gateB, nsqA, nsqB);
    // K3: quantile thresholds
    kZeroHist<<<(2 * 65536) / 256, 256, 0, stream>>>(hist);
    kHist<<<kNP / 256, 256, 0, stream>>>(nsqA, nsqB, hist);
    kQuantile<<<1, 256, 0, stream>>>(hist, thr);
    // K4: masked permutes to GEMM layouts
    kPermute<<<dim3(16, 4, 512), dim3(32, 8), 0, stream>>>(
        aB, a_t, nsqA, thr + 0, /*inOffMulZ=*/kN * kD, /*inRowStride=*/kD,
        /*pMulR=*/1, /*pMulZ=*/kN);
    kPermute<<<dim3(16, 4, 512), dim3(32, 8), 0, stream>>>(
        bB, b_t, nsqB, thr + 1, /*inOffMulZ=*/kD, /*inRowStride=*/kN * kD,
        /*pMulR=*/kN, /*pMulZ=*/1);
    // K5: triangle GEMMs (128 x 512x512x512 bf16 WMMA)
    kTriangle<<<dim3(kN / 16, kN / 16 / 8, kD), 256, 0, stream>>>(a_t, b_t, tF);
    // K6: LN(t) -> bf16
    kLayerNorm<<<kNP / 8, 256, 0, stream>>>(tF, ln_out_g, ln_out_b, tln);
    // K7: gated projection + residual
    kFinal<<<kNP / 128, 256, 0, stream>>>(tln, wF + 5 * kD * kD, gateB, z, out);
}
